// NanoSSSM_9174050144315
// MI455X (gfx1250) — matrix operand order repaired and checked by reading, not compiled, not run on hardware
//
#include <hip/hip_runtime.h>
#include <math.h>

// MI455X / gfx1250, wave32. WMMA f32<-f16 16x16x32 GEMM pipeline for NanoSSSM,
// with CDNA5 async global->LDS tile staging and a chunked parallel scan.

typedef __attribute__((ext_vector_type(16))) _Float16 v16h;
typedef __attribute__((ext_vector_type(8)))  _Float16 v8h;
typedef __attribute__((ext_vector_type(8)))  float    v8f;

#define WAVE 32
#define BK   32
#define LPAD 8   // halves of padding per LDS row (16B) to spread banks

__device__ __forceinline__ float sigmoidf_(float x) { return 1.0f / (1.0f + expf(-x)); }
__device__ __forceinline__ float softplusf_(float x) {
    if (x > 20.0f) return x;
    return log1pf(expf(x));
}
// Wave-relative LDS byte address of a __shared__ object: per the CDNA5
// aperture rules, flat-address bits [31:0] ARE the LDS offset.
__device__ __forceinline__ unsigned lds_addr32_(const void* p) {
    return (unsigned)(unsigned long long)p;
}

// ---------------------------------------------------------------------------
// Generic WMMA GEMM:  Out = A(MxK, f16, row-major) * Bw(NxK, f16, row-major)^T
// Block tile BM x BN, K-step 32, 256 threads = 8 waves arranged WGM x WGN.
// Tiles are staged with GLOBAL_LOAD_ASYNC_TO_LDS_B128 (ASYNCcnt), double-
// buffered. EPI selects the fused epilogue.
// ---------------------------------------------------------------------------
template <int BM, int BN, int WGM, int WGN, int EPI>
__global__ __launch_bounds__(256) void wmma_gemm_kernel(
    const _Float16* __restrict__ A, const _Float16* __restrict__ Bw,
    int M, int N, int K,
    const float* __restrict__ ep0, const float* __restrict__ ep1,
    _Float16* __restrict__ o16a, _Float16* __restrict__ o16b,
    float* __restrict__ of32a, float* __restrict__ of32b)
{
    constexpr int LDT = BK + LPAD;          // LDS row stride (halves)
    constexpr int WTM = BM / WGM;           // wave tile M
    constexpr int WTN = BN / WGN;           // wave tile N
    constexpr int MT  = WTM / 16;           // 16x16 tiles per wave (M)
    constexpr int NT  = WTN / 16;           // 16x16 tiles per wave (N)
    constexpr int ACH = (BM * BK) / (8 * 256);  // 16B chunks per thread (A)
    constexpr int BCH = (BN * BK) / (8 * 256);  // 16B chunks per thread (B)

    __shared__ alignas(16) _Float16 As[2][BM * LDT];
    __shared__ alignas(16) _Float16 Bs[2][BN * LDT];

    const int tid = threadIdx.x;
    const int wid = tid / WAVE;
    const int lid = tid % WAVE;
    const int g   = lid >> 4;     // lane group (0/1)
    const int r   = lid & 15;     // lane row/col within 16
    const int waveM = wid / WGN;
    const int waveN = wid % WGN;
    const long blockM = (long)blockIdx.y * BM;
    const long blockN = (long)blockIdx.x * BN;

    const int nk = K / BK;

    v8f acc[MT][NT];
#pragma unroll
    for (int mt = 0; mt < MT; ++mt)
#pragma unroll
        for (int nt = 0; nt < NT; ++nt)
#pragma unroll
            for (int j = 0; j < 8; ++j) acc[mt][nt][j] = 0.0f;

    // Async global->LDS staging of one BMxBK (and BNxBK) tile, no VGPR data.
    auto issue_async = [&](int kk, int buf) {
#pragma unroll
        for (int c = 0; c < ACH; ++c) {
            int lin = c * 256 + tid;
            int row = lin >> 2;                 // BK/8 = 4 chunks per row
            int kc  = (lin & 3) * 8;
            unsigned ldst = lds_addr32_(&As[buf][row * LDT + kc]);
            unsigned long long ga =
                (unsigned long long)(A + (blockM + row) * (long)K + (long)kk * BK + kc);
            asm volatile("global_load_async_to_lds_b128 %0, %1, off"
                         :: "v"(ldst), "v"(ga) : "memory");
        }
#pragma unroll
        for (int c = 0; c < BCH; ++c) {
            int lin = c * 256 + tid;
            int row = lin >> 2;
            int kc  = (lin & 3) * 8;
            unsigned ldst = lds_addr32_(&Bs[buf][row * LDT + kc]);
            unsigned long long ga =
                (unsigned long long)(Bw + (blockN + row) * (long)K + (long)kk * BK + kc);
            asm volatile("global_load_async_to_lds_b128 %0, %1, off"
                         :: "v"(ldst), "v"(ga) : "memory");
        }
    };

    auto compute = [&](int buf) {
        // A fragment per ISA 7.12.2: lane holds row M=r; element i -> K =
        //   (i<8 ? g*8+i : 16+g*8+(i-8))  => two contiguous 8-half chunks.
        v16h af[MT];
#pragma unroll
        for (int mt = 0; mt < MT; ++mt) {
            const _Float16* p = &As[buf][(waveM * WTM + mt * 16 + r) * LDT + g * 8];
            v8h lo = *(const v8h*)p;
            v8h hi = *(const v8h*)(p + 16);
#pragma unroll
            for (int i = 0; i < 8; ++i) { af[mt][i] = lo[i]; af[mt][i + 8] = hi[i]; }
        }
        // B fragment: lane holds column N=r; element i -> K =
        //   (i<8 ? g*8+i : 16+g*8+(i-8))  => two contiguous 8-half chunks.
#pragma unroll
        for (int nt = 0; nt < NT; ++nt) {
            const _Float16* p = &Bs[buf][(waveN * WTN + nt * 16 + r) * LDT + g * 8];
            v8h lo = *(const v8h*)p;
            v8h hi = *(const v8h*)(p + 16);
            v16h bf;
#pragma unroll
            for (int i = 0; i < 8; ++i) { bf[i] = lo[i]; bf[i + 8] = hi[i]; }
#pragma unroll
            for (int mt = 0; mt < MT; ++mt) {
                acc[mt][nt] = __builtin_amdgcn_wmma_f32_16x16x32_f16(
                    false, af[mt], false, bf, (short)0, acc[mt][nt], false, false);
            }
        }
    };

    issue_async(0, 0);
    asm volatile("s_wait_asynccnt 0x0" ::: "memory");
    __syncthreads();
    for (int kk = 0; kk < nk; ++kk) {
        int cur = kk & 1;
        if (kk + 1 < nk) issue_async(kk + 1, 1 - cur);  // writes buffer 1-cur
        compute(cur);                                   // reads buffer cur
        if (kk + 1 < nk) {
            asm volatile("s_wait_asynccnt 0x0" ::: "memory");
            __syncthreads();
        }
    }

    // Fused epilogues. D element j lives at row = g*8 + j, col = r of its tile.
#pragma unroll
    for (int mt = 0; mt < MT; ++mt)
#pragma unroll
        for (int nt = 0; nt < NT; ++nt)
#pragma unroll
            for (int j = 0; j < 8; ++j) {
                long row = blockM + waveM * WTM + mt * 16 + g * 8 + j;
                int  col = (int)(blockN + waveN * WTN + nt * 16 + r);
                float v  = acc[mt][nt][j];
                if constexpr (EPI == 0) {
                    // in_proj: cols [0,768) -> xc = silu(v*conv_w+conv_b) (f16)
                    //          cols [768,1536) -> z raw (f16)
                    if (col < 768) {
                        float c = v * ep0[col] + ep1[col];
                        o16a[row * 768 + col] = (_Float16)(c * sigmoidf_(c));
                    } else {
                        o16b[row * 768 + (col - 768)] = (_Float16)v;
                    }
                } else if constexpr (EPI == 1) {
                    // x_proj (N padded to 64): cols [0,48) -> delta_raw (f16,
                    // zero-padded to 64 wide); col 48 -> B; col 49 -> C.
                    o16a[row * 64 + col] = (col < 48) ? (_Float16)v : (_Float16)0.0f;
                    if (col == 48) of32a[row] = v;
                    if (col == 49) of32b[row] = v;
                } else if constexpr (EPI == 2) {
                    // dt_proj: delta = softplus(v + dt_proj_b) (f32)
                    of32a[row * (long)N + col] = softplusf_(v + ep1[col]);
                } else {
                    // out_proj: plain f32 store
                    of32a[row * (long)N + col] = v;
                }
            }
}

// ---------------------------------------------------------------------------
// RMSNorm: one row (b,l) per block, 256 threads, writes f16.
// ---------------------------------------------------------------------------
__global__ __launch_bounds__(256) void rmsnorm_kernel(
    const float* __restrict__ x, const float* __restrict__ w,
    _Float16* __restrict__ xn, int D)
{
    long row = blockIdx.x;
    const float* xr = x + row * (long)D;
    float s = 0.0f;
    for (int i = threadIdx.x; i < D; i += 256) { float v = xr[i]; s += v * v; }
#pragma unroll
    for (int o = 16; o > 0; o >>= 1) s += __shfl_down(s, o, 32);
    __shared__ float red[8];
    __shared__ float scale;
    if ((threadIdx.x & 31) == 0) red[threadIdx.x >> 5] = s;
    __syncthreads();
    if (threadIdx.x == 0) {
        float t = 0.0f;
        for (int i = 0; i < 8; ++i) t += red[i];
        scale = rsqrtf(t / (float)D + 1e-6f);
    }
    __syncthreads();
    float sc = scale;
    for (int i = threadIdx.x; i < D; i += 256)
        xn[row * (long)D + i] = (_Float16)(xr[i] * sc * w[i]);
}

// ---------------------------------------------------------------------------
// Weight conversion / padding
// ---------------------------------------------------------------------------
__global__ void cvt_f32_f16_kernel(const float* __restrict__ s, _Float16* __restrict__ d, long n)
{
    long i = (long)blockIdx.x * 256 + threadIdx.x;
    if (i < n) d[i] = (_Float16)s[i];
}

// x_proj_w (50x768) -> (64x768) f16, rows >=50 zero
__global__ void pad_xproj_kernel(const float* __restrict__ s, _Float16* __restrict__ d)
{
    int i = blockIdx.x * 256 + threadIdx.x;
    if (i < 64 * 768) {
        int n = i / 768, k = i % 768;
        d[i] = (n < 50) ? (_Float16)s[n * 768 + k] : (_Float16)0.0f;
    }
}

// dt_proj_w (768x48) -> (768x64) f16, k >= 48 zero
__global__ void pad_dtw_kernel(const float* __restrict__ s, _Float16* __restrict__ d)
{
    int i = blockIdx.x * 256 + threadIdx.x;
    if (i < 768 * 64) {
        int e = i / 64, k = i % 64;
        d[i] = (k < 48) ? (_Float16)s[e * 48 + k] : (_Float16)0.0f;
    }
}

// ---------------------------------------------------------------------------
// Chunked parallel scan (N_STATE==1). h_t = a_t h_{t-1} + u_t with
// a = exp(delta*A), u = delta*B*xc. A chunk is summarized by (P, S):
// h_out = P*h_in + S, composed as (P1P2, a2*S1+S2). L=4096 -> 32 chunks x 128.
// Thread t = c*6144 + (b*768+e): consecutive threads differ in e => coalesced.
// ---------------------------------------------------------------------------
#define NBE   (8 * 768)      // 6144 channels
#define NCHK  32
#define CLEN  128            // 4096 / 32

__global__ __launch_bounds__(256) void scan_chunk_kernel(
    const float* __restrict__ delta, const _Float16* __restrict__ xc16,
    const float* __restrict__ Bm, const float* __restrict__ A_log,
    float* __restrict__ Pc, float* __restrict__ Sc)
{
    int t = blockIdx.x * 256 + threadIdx.x;
    if (t >= NCHK * NBE) return;
    int c = t / NBE, be = t % NBE;
    int b = be / 768, e = be % 768;
    float Ae = -expf(A_log[e]);   // A_log shape (768,1)
    float P = 1.0f, S = 0.0f;
    long l0 = (long)b * 4096 + (long)c * CLEN;
    for (int i = 0; i < CLEN; ++i) {
        long rowIdx = l0 + i;
        long idx    = rowIdx * 768 + e;
        float dt = delta[idx];
        float a  = expf(dt * Ae);
        float u  = dt * Bm[rowIdx] * (float)xc16[idx];
        P *= a;
        S  = a * S + u;
    }
    Pc[t] = P;
    Sc[t] = S;
}

__global__ __launch_bounds__(256) void scan_prefix_kernel(
    const float* __restrict__ Pc, const float* __restrict__ Sc,
    float* __restrict__ Hin)
{
    int be = blockIdx.x * 256 + threadIdx.x;
    if (be >= NBE) return;
    float h = 0.0f;
    for (int c = 0; c < NCHK; ++c) {
        int t = c * NBE + be;
        Hin[t] = h;
        h = Pc[t] * h + Sc[t];
    }
}

__global__ __launch_bounds__(256) void scan_emit_kernel(
    const float* __restrict__ delta, const _Float16* __restrict__ xc16,
    const _Float16* __restrict__ z16, const float* __restrict__ Bm,
    const float* __restrict__ Cm, const float* __restrict__ A_log,
    const float* __restrict__ Dp, const float* __restrict__ Hin,
    _Float16* __restrict__ y16)
{
    int t = blockIdx.x * 256 + threadIdx.x;
    if (t >= NCHK * NBE) return;
    int c = t / NBE, be = t % NBE;
    int b = be / 768, e = be % 768;
    float Ae = -expf(A_log[e]);
    float De = Dp[e];
    float h  = Hin[t];
    long l0 = (long)b * 4096 + (long)c * CLEN;
    for (int i = 0; i < CLEN; ++i) {
        long rowIdx = l0 + i;
        long idx    = rowIdx * 768 + e;
        float dt = delta[idx];
        float xc = (float)xc16[idx];
        float a  = expf(dt * Ae);
        h = a * h + dt * Bm[rowIdx] * xc;
        float y = h * Cm[rowIdx] + De * xc;
        float z = (float)z16[idx];
        y *= z * sigmoidf_(z);
        y16[idx] = (_Float16)y;
    }
}

// ---------------------------------------------------------------------------
extern "C" void kernel_launch(void* const* d_in, const int* in_sizes, int n_in,
                              void* d_out, int out_size, void* d_ws, size_t ws_size,
                              hipStream_t stream)
{
    const float* x         = (const float*)d_in[0];
    const float* rms_w     = (const float*)d_in[1];
    const float* in_proj_w = (const float*)d_in[2];
    const float* conv_w    = (const float*)d_in[3];
    const float* conv_b    = (const float*)d_in[4];
    const float* x_proj_w  = (const float*)d_in[5];
    const float* dt_proj_w = (const float*)d_in[6];
    const float* dt_proj_b = (const float*)d_in[7];
    const float* A_log     = (const float*)d_in[8];
    const float* Dp        = (const float*)d_in[9];
    const float* out_w     = (const float*)d_in[10];
    float* out = (float*)d_out;

    const long Mrows = 8L * 4096;   // 32768
    const int  Dm = 768, N1 = 1536;

    char* wsp = (char*)d_ws;
    size_t off = 0;
    auto wsalloc = [&](size_t bytes) -> void* {
        void* p = wsp + off;
        off = (off + bytes + 255) & ~(size_t)255;
        return p;
    };
    _Float16* w1_16  = (_Float16*)wsalloc(1536L * 768 * 2);
    _Float16* wx_16  = (_Float16*)wsalloc(64L * 768 * 2);
    _Float16* wdt_16 = (_Float16*)wsalloc(768L * 64 * 2);
    _Float16* wo_16  = (_Float16*)wsalloc(768L * 768 * 2);
    _Float16* xn16   = (_Float16*)wsalloc(Mrows * 768 * 2);
    _Float16* xc16   = (_Float16*)wsalloc(Mrows * 768 * 2);
    _Float16* z16    = (_Float16*)wsalloc(Mrows * 768 * 2);
    _Float16* dtA16  = (_Float16*)wsalloc(Mrows * 64 * 2);
    float*    Bmf    = (float*)wsalloc(Mrows * 4);
    float*    Cmf    = (float*)wsalloc(Mrows * 4);
    float*    deltaf = (float*)wsalloc(Mrows * 768 * 4);
    _Float16* y16    = (_Float16*)wsalloc(Mrows * 768 * 2);
    float*    Pc     = (float*)wsalloc((long)NCHK * NBE * 4);
    float*    Sc     = (float*)wsalloc((long)NCHK * NBE * 4);
    float*    Hin    = (float*)wsalloc((long)NCHK * NBE * 4);

    // Weight conversions / padding (tiny)
    cvt_f32_f16_kernel<<<(int)((1536L * 768 + 255) / 256), 256, 0, stream>>>(in_proj_w, w1_16, 1536L * 768);
    cvt_f32_f16_kernel<<<(int)((768L * 768 + 255) / 256), 256, 0, stream>>>(out_w, wo_16, 768L * 768);
    pad_xproj_kernel<<<(64 * 768 + 255) / 256, 256, 0, stream>>>(x_proj_w, wx_16);
    pad_dtw_kernel<<<(768 * 64 + 255) / 256, 256, 0, stream>>>(dt_proj_w, wdt_16);

    // RMSNorm -> xn (f16)
    rmsnorm_kernel<<<(int)Mrows, 256, 0, stream>>>(x, rms_w, xn16, Dm);

    // GEMM1: xz = xn @ in_proj_w^T ; fused split + conv*silu
    {
        dim3 grid(N1 / 128, (int)(Mrows / 128));
        wmma_gemm_kernel<128, 128, 2, 4, 0><<<grid, 256, 0, stream>>>(
            xn16, w1_16, (int)Mrows, N1, 768,
            conv_w, conv_b, xc16, z16, nullptr, nullptr);
    }
    // GEMM2: dbc = xc @ x_proj_w^T (N padded to 64); split delta_raw / B / C
    {
        dim3 grid(1, (int)(Mrows / 128));
        wmma_gemm_kernel<128, 64, 4, 2, 1><<<grid, 256, 0, stream>>>(
            xc16, wx_16, (int)Mrows, 64, 768,
            nullptr, nullptr, dtA16, nullptr, Bmf, Cmf);
    }
    // GEMM3: delta = softplus(delta_raw @ dt_proj_w^T + dt_proj_b)
    {
        dim3 grid(768 / 128, (int)(Mrows / 128));
        wmma_gemm_kernel<128, 128, 2, 4, 2><<<grid, 256, 0, stream>>>(
            dtA16, wdt_16, (int)Mrows, 768, 64,
            nullptr, dt_proj_b, nullptr, nullptr, deltaf, nullptr);
    }
    // Chunked selective scan + gating -> y (f16)
    scan_chunk_kernel<<<(NCHK * NBE) / 256, 256, 0, stream>>>(
        deltaf, xc16, Bmf, A_log, Pc, Sc);
    scan_prefix_kernel<<<NBE / 256, 256, 0, stream>>>(Pc, Sc, Hin);
    scan_emit_kernel<<<(NCHK * NBE) / 256, 256, 0, stream>>>(
        deltaf, xc16, z16, Bmf, Cmf, A_log, Dp, Hin, y16);
    // GEMM4: out = y @ out_proj_w^T (f32)
    {
        dim3 grid(768 / 128, (int)(Mrows / 128));
        wmma_gemm_kernel<128, 128, 2, 4, 3><<<grid, 256, 0, stream>>>(
            y16, wo_16, (int)Mrows, 768, 768,
            nullptr, nullptr, nullptr, nullptr, out, nullptr);
    }
}
